// NODE_64536178590129
// MI455X (gfx1250) — matrix operand order repaired and checked by reading, not compiled, not run on hardware
//
#include <hip/hip_runtime.h>

// ---------------------------------------------------------------------------
// ODE-RNN (RK4 liquid net) for MI455X / gfx1250.
// B=32768, IN=512, H=1024, 6 unfolds x 4 RK4 stages.
// Compute-bound (1.68 TFLOP vs ~320MB HBM) -> bf16 WMMA w/ f32 accumulate.
// Per-row recurrence -> persistent workgroup owns 32 rows, runs all 24 GEMM
// stages locally; Wh streamed from L2 each stage in pre-swizzled bf16 form.
// ---------------------------------------------------------------------------

typedef __attribute__((ext_vector_type(16))) __bf16     v16bf;
typedef __attribute__((ext_vector_type(8)))  float      v8f;
typedef __attribute__((ext_vector_type(4)))  unsigned   v4u;

#define DEVFN __device__ __forceinline__

constexpr int   Bsz      = 32768;
constexpr int   INs      = 512;
constexpr int   Hs       = 1024;
constexpr int   MBLK     = 32;        // rows per workgroup
constexpr int   NTHREADS = 512;       // 16 waves
constexpr int   AROW     = Hs * 2 + 16; // 2064 B: padded bf16 row (conflict-free)
constexpr float DT       = 0.1f;

DEVFN unsigned short f2bf(float f) {
  union { float f; unsigned u; } z; z.f = f;
  unsigned r = z.u + 0x7FFFu + ((z.u >> 16) & 1u);   // round-to-nearest-even
  return (unsigned short)(r >> 16);
}

DEVFN float fast_tanh(float x) {
#if __has_builtin(__builtin_amdgcn_tanhf)
  return __builtin_amdgcn_tanhf(x);
#elif __has_builtin(__builtin_amdgcn_tanh_f32)
  return __builtin_amdgcn_tanh_f32(x);
#else
  float ax = __builtin_fabsf(x);
  float e  = __expf(-2.0f * ax);
  float t  = 1.0f - 2.0f * e / (1.0f + e);
  return __builtin_copysignf(t, x);
#endif
}

// ---------------------------------------------------------------------------
// Pre-swizzle W (f32, [1024 x 1536]) into bf16 B-fragment order:
//   layout [nt][kc][c][lane][e], nt=N-tile(16 cols), kc=K-chunk(32),
//   lane holds column (lane&15) with K-half selected by lane>=16,
//   c selects the 8-element (16B) subchunk -> every wave B-load is one
//   contiguous 512B global_load_b128 group.
// ---------------------------------------------------------------------------
__global__ void prep_weights(const float* __restrict__ W,
                             unsigned short* __restrict__ wx,
                             unsigned short* __restrict__ wh) {
  const int NX = 64 * 16 * 2 * 32 * 8;   // Wx: K=512  (524288)
  const int NH = 64 * 32 * 2 * 32 * 8;   // Wh: K=1024 (1048576)
  int idx = blockIdx.x * blockDim.x + threadIdx.x;
  if (idx < NX) {
    int e = idx & 7, t = idx >> 3;
    int lane = t & 31; t >>= 5;
    int c = t & 1;     t >>= 1;
    int kc = t & 15;   t >>= 4;
    int nt = t;
    int row = nt * 16 + (lane & 15);
    int k   = kc * 32 + ((lane < 16) ? 0 : 8) + c * 16 + e;
    wx[idx] = f2bf(W[(size_t)row * 1536 + k]);
  } else if (idx < NX + NH) {
    int j = idx - NX;
    int e = j & 7, t = j >> 3;
    int lane = t & 31; t >>= 5;
    int c = t & 1;     t >>= 1;
    int kc = t & 31;   t >>= 5;
    int nt = t;
    int row = nt * 16 + (lane & 15);
    int k   = kc * 32 + ((lane < 16) ? 0 : 8) + c * 16 + e;
    wh[j] = f2bf(W[(size_t)row * 1536 + 512 + k]);
  }
}

// ---------------------------------------------------------------------------
// One GEMM pass: acc[mi][j] += A(sA rows) x B(wswz), K = KC*32.
// A fragments from LDS (ISA 16-bit A layout: lane<16 -> K{0-7,16-23},
// lane>=16 -> K{8-15,24-31}); B fragments from pre-swizzled global.
// ---------------------------------------------------------------------------
template <int KC>
DEVFN void gemm_tiles(const unsigned char* sA,
                      const unsigned short* __restrict__ wswz,
                      int w, int lane, v8f acc[2][4]) {
  const int lh    = lane & 15;
  const int koff8 = (lane < 16) ? 0 : 8;
  for (int kc = 0; kc < KC; ++kc) {
    v16bf a[2];
#pragma unroll
    for (int mi = 0; mi < 2; ++mi) {
      const unsigned char* p = sA + (mi * 16 + lh) * AROW + (kc * 32 + koff8) * 2;
      union { v4u q[2]; v16bf v; } ua;
      ua.q[0] = *(const v4u*)p;          // ds_load_b128: K base chunk
      ua.q[1] = *(const v4u*)(p + 32);   // ds_load_b128: K base+16 chunk
      a[mi] = ua.v;
    }
#pragma unroll
    for (int j = 0; j < 4; ++j) {
      const unsigned short* bp = wswz + ((size_t)((w * 4 + j) * KC + kc)) * 512 + lane * 8;
      union { v4u q[2]; v16bf v; } ub;
      ub.q[0] = *(const v4u*)bp;           // global_load_b128 (coalesced 512B/wave)
      ub.q[1] = *(const v4u*)(bp + 256);   // global_load_b128
      v16bf b = ub.v;
      acc[0][j] = __builtin_amdgcn_wmma_f32_16x16x32_bf16(
          false, a[0], false, b, (short)0, acc[0][j], false, false);
      acc[1][j] = __builtin_amdgcn_wmma_f32_16x16x32_bf16(
          false, a[1], false, b, (short)0, acc[1][j], false, false);
    }
  }
}

// ---------------------------------------------------------------------------
// Persistent RK4 kernel. Each block: 32 rows. Wave w owns N-tiles 4w..4w+3
// for both 16-row M-tiles; h / rk / xp held in registers in the native
// C/D fragment layout (row = v + 8*(lane>=16), col = lane&15).
// ---------------------------------------------------------------------------
__global__ __launch_bounds__(NTHREADS, 1)
void node_rk4(const float* __restrict__ inputs, const float* __restrict__ state,
              const float* __restrict__ bias,
              const unsigned short* __restrict__ wx,
              const unsigned short* __restrict__ wh,
              float* __restrict__ out) {
  extern __shared__ unsigned char sA[];   // MBLK * AROW = 66048 B (dynamic LDS)

  const int tid  = threadIdx.x;
  const int lane = tid & 31;
  const int w    = tid >> 5;      // wave 0..15
  const int lh   = lane & 15;
  const int hi   = lane >> 4;     // 0/1 -> row half of tile
  const int r0   = blockIdx.x * MBLK;

  float h [2][4][8];
  float rk[2][4][8];
  float xp[2][4][8];

  // ---- init h from state (owned-fragment layout) ----
#pragma unroll
  for (int mi = 0; mi < 2; ++mi)
#pragma unroll
    for (int j = 0; j < 4; ++j) {
      const int col = (w * 4 + j) * 16 + lh;
#pragma unroll
      for (int v = 0; v < 8; ++v)
        h[mi][j][v] = state[(size_t)(r0 + mi * 16 + 8 * hi + v) * Hs + col];
    }

  // ---- stage input block into LDS as bf16 (A layout) ----
  for (int t = tid; t < MBLK * (INs / 2); t += NTHREADS) {
    int row = t >> 8;            // 256 packed pairs per row
    int k2  = (t & 255) * 2;
    const float* src = inputs + (size_t)(r0 + row) * INs + k2;
    unsigned pack = (unsigned)f2bf(src[0]) | ((unsigned)f2bf(src[1]) << 16);
    *(unsigned*)(sA + row * AROW + k2 * 2) = pack;
  }
  __syncthreads();

  v8f acc[2][4];

  // ---- xp = inputs @ Wx^T + b ----
#pragma unroll
  for (int mi = 0; mi < 2; ++mi)
#pragma unroll
    for (int j = 0; j < 4; ++j)
#pragma unroll
      for (int v = 0; v < 8; ++v) acc[mi][j][v] = 0.0f;

  gemm_tiles<16>(sA, wx, w, lane, acc);

#pragma unroll
  for (int mi = 0; mi < 2; ++mi)
#pragma unroll
    for (int j = 0; j < 4; ++j) {
      float bc = bias[(w * 4 + j) * 16 + lh];
#pragma unroll
      for (int v = 0; v < 8; ++v) xp[mi][j][v] = acc[mi][j][v] + bc;
    }

  __syncthreads();               // all waves done reading staged inputs
  // ---- h_eff for first stage = h ----
#pragma unroll
  for (int mi = 0; mi < 2; ++mi)
#pragma unroll
    for (int j = 0; j < 4; ++j) {
      const int col = (w * 4 + j) * 16 + lh;
#pragma unroll
      for (int v = 0; v < 8; ++v)
        *(unsigned short*)(sA + (mi * 16 + 8 * hi + v) * AROW + col * 2) =
            f2bf(h[mi][j][v]);
    }
  __syncthreads();

  // ---- 6 unfolds x 4 RK4 stages, all local to the block ----
  for (int u = 0; u < 6; ++u) {
#pragma unroll
    for (int st = 0; st < 4; ++st) {
#pragma unroll
      for (int mi = 0; mi < 2; ++mi)
#pragma unroll
        for (int j = 0; j < 4; ++j)
#pragma unroll
          for (int v = 0; v < 8; ++v) acc[mi][j][v] = 0.0f;

      gemm_tiles<32>(sA, wh, w, lane, acc);

      __syncthreads();           // all reads of sA complete before rewrite

      const float cs = (st < 2) ? 0.5f : 1.0f;          // h_eff coefficient
      const float wk = (st == 0 || st == 3) ? 1.0f : 2.0f; // RK weight
#pragma unroll
      for (int mi = 0; mi < 2; ++mi)
#pragma unroll
        for (int j = 0; j < 4; ++j) {
          const int col = (w * 4 + j) * 16 + lh;
#pragma unroll
          for (int v = 0; v < 8; ++v) {
            float f = fast_tanh(xp[mi][j][v] + acc[mi][j][v]);
            float k = DT * f;
            float r = (st == 0) ? k : (rk[mi][j][v] + wk * k);
            rk[mi][j][v] = r;
            float he;
            if (st < 3) {
              he = h[mi][j][v] + cs * k;
            } else {
              he = h[mi][j][v] + r * (1.0f / 6.0f);
              h[mi][j][v] = he;
            }
            *(unsigned short*)(sA + (mi * 16 + 8 * hi + v) * AROW + col * 2) =
                f2bf(he);
          }
        }
      __syncthreads();           // staged h_eff visible to all waves
    }
  }

  // ---- write final state ----
#pragma unroll
  for (int mi = 0; mi < 2; ++mi)
#pragma unroll
    for (int j = 0; j < 4; ++j) {
      const int col = (w * 4 + j) * 16 + lh;
#pragma unroll
      for (int v = 0; v < 8; ++v)
        out[(size_t)(r0 + mi * 16 + 8 * hi + v) * Hs + col] = h[mi][j][v];
    }
}

// ---------------------------------------------------------------------------
extern "C" void kernel_launch(void* const* d_in, const int* in_sizes, int n_in,
                              void* d_out, int out_size, void* d_ws, size_t ws_size,
                              hipStream_t stream) {
  const float* inputs = (const float*)d_in[0];   // [32768, 512]
  const float* state  = (const float*)d_in[1];   // [32768, 1024]
  const float* W      = (const float*)d_in[2];   // [1024, 1536]
  const float* b      = (const float*)d_in[3];   // [1024]
  float* out = (float*)d_out;

  unsigned short* wx = (unsigned short*)d_ws;        // 512K bf16 = 1 MB
  unsigned short* wh = wx + 512 * 1024;              // 1M bf16  = 2 MB

  const int nprep = 64 * 16 * 2 * 32 * 8 + 64 * 32 * 2 * 32 * 8; // 1572864
  prep_weights<<<nprep / 256, 256, 0, stream>>>(W, wx, wh);

  node_rk4<<<Bsz / MBLK, NTHREADS, MBLK * AROW, stream>>>(
      inputs, state, b, wx, wh, out);
}
